// PointResNet_TorchSparse_59880434041080
// MI455X (gfx1250) — matrix operand order repaired and checked by reading, not compiled, not run on hardware
//
#include <hip/hip_runtime.h>
#include <hip/hip_bf16.h>
#include <math.h>

typedef __attribute__((ext_vector_type(16))) _Float16 v16h;
typedef __attribute__((ext_vector_type(8)))  _Float16 v8h;
typedef __attribute__((ext_vector_type(8)))  float    v8f;

#define NVOX  131072
#define NPERB 65536
#define EPSV  1e-5f

// ---------------------------------------------------------------------------
// Repack W (fp32 [27][Cin][Cout]) -> fp16 in WMMA-B per-lane layout.
// Tile (k,kt,nt) = 32x16 slab of W[k]; 512 halves per tile:
//   pw[tile*512 + lane*16 + h] = W[k][kt*32 + (lane>>4)*16 + h][nt*16 + (lane&15)]
// (zero-padded when kin >= Cin; handles Cin=3 -> CINP=32)
// grid = (27, KT, NTt), block = 512
__global__ void repack_w_kernel(const float* __restrict__ W, _Float16* __restrict__ pw,
                                int Cin, int Cout)
{
  const int k = blockIdx.x, kt = blockIdx.y, nt = blockIdx.z;
  const int KT = gridDim.y, NTt = gridDim.z;
  const int t    = threadIdx.x;      // 0..511
  const int lane = t >> 4;
  const int h    = t & 15;
  const int kin  = kt * 32 + (lane >> 4) * 16 + h;
  const int n    = nt * 16 + (lane & 15);
  float v = (kin < Cin) ? W[((size_t)k * Cin + kin) * Cout + n] : 0.0f;
  pw[((size_t)(k * KT + kt) * NTt + nt) * 512 + t] = (_Float16)v;
}

// feats fp32 [N,3] -> fp16 [N+1, 32] (channels padded, row N = zero pad row)
__global__ void feats_prep_kernel(const float* __restrict__ feats, _Float16* __restrict__ fh)
{
  int idx = blockIdx.x * blockDim.x + threadIdx.x;
  if (idx >= (NVOX + 1) * 32) return;
  int i = idx >> 5, c = idx & 31;
  float v = (i < NVOX && c < 3) ? feats[i * 3 + c] : 0.0f;
  fh[idx] = (_Float16)v;
}

// ---------------------------------------------------------------------------
// Submanifold conv: out[i,:] = sum_{k<27} fh[neigh[i,k]] @ Wk  via WMMA f16.
// One wave: 32 rows x 64 cols (2 M-tiles x 4 N-tiles), fp32 accumulators.
// grid = (NVOX/(32*8), COUT/64), block = 256 (8 waves).
template<int CINP, int COUT>
__global__ __launch_bounds__(256) void conv_wmma_kernel(
    const _Float16* __restrict__ fh,   // [>=NVOX+1, CINP] fp16, row NVOX zero
    const _Float16* __restrict__ pw,   // packed weights
    const int*      __restrict__ neigh,// [NVOX, 27]
    float*          __restrict__ out)  // [NVOX, COUT]
{
  constexpr int KT  = CINP / 32;
  constexpr int NTt = COUT / 16;
  const int lane = threadIdx.x & 31;
  const int wave = threadIdx.x >> 5;
  const int r    = lane & 15;           // row within tile / output column lane
  const int ksel = lane >> 4;           // K-half select for A; row-half for C/D
  const int pair = blockIdx.x * 8 + wave;
  const int rowA = pair * 32 + r;       // M-tile 0
  const int rowB = rowA + 16;           // M-tile 1
  const int colgrp = blockIdx.y;        // 64 output columns per group

  v8f acc[2][4];
  #pragma unroll
  for (int m = 0; m < 2; ++m)
    #pragma unroll
    for (int t = 0; t < 4; ++t)
      acc[m][t] = v8f{};

  const _Float16* pwl = pw + (size_t)(colgrp * 4) * 512 + (size_t)lane * 16;
  const int* nA = neigh + (size_t)rowA * 27;
  const int* nB = neigh + (size_t)rowB * 27;

  #pragma unroll 1
  for (int k = 0; k < 27; ++k) {
    const int ia = nA[k];
    const int ib = nB[k];
    const _Float16* pa = fh + (size_t)ia * CINP + ksel * 16;
    const _Float16* pb = fh + (size_t)ib * CINP + ksel * 16;
    const _Float16* bk = pwl + (size_t)k * (KT * NTt) * 512;
    #pragma unroll
    for (int kt = 0; kt < KT; ++kt) {
      v8h a0lo = *(const v8h*)(pa + kt * 32);
      v8h a0hi = *(const v8h*)(pa + kt * 32 + 8);
      v8h a1lo = *(const v8h*)(pb + kt * 32);
      v8h a1hi = *(const v8h*)(pb + kt * 32 + 8);
      v16h a0 = __builtin_shufflevector(a0lo, a0hi, 0,1,2,3,4,5,6,7,8,9,10,11,12,13,14,15);
      v16h a1 = __builtin_shufflevector(a1lo, a1hi, 0,1,2,3,4,5,6,7,8,9,10,11,12,13,14,15);
      const _Float16* bkt = bk + (size_t)kt * NTt * 512;
      #pragma unroll
      for (int t = 0; t < 4; ++t) {
        v16h b = *(const v16h*)(bkt + t * 512);
        acc[0][t] = __builtin_amdgcn_wmma_f32_16x16x32_f16(
            false, a0, false, b, (short)0, acc[0][t], false, false);
        acc[1][t] = __builtin_amdgcn_wmma_f32_16x16x32_f16(
            false, a1, false, b, (short)0, acc[1][t], false, false);
      }
    }
  }

  // C/D layout: VGPR v -> row = base + ksel*8 + v, col = lane&15
  const int col = colgrp * 64 + r;
  #pragma unroll
  for (int m = 0; m < 2; ++m) {
    const int rbase = pair * 32 + m * 16 + ksel * 8;
    #pragma unroll
    for (int t = 0; t < 4; ++t) {
      float* op = out + (size_t)rbase * COUT + col + t * 16;
      #pragma unroll
      for (int v = 0; v < 8; ++v)
        op[(size_t)v * COUT] = acc[m][t][v];
    }
  }
}

// ---------------------------------------------------------------------------
// Deterministic per-(batch,channel) sum / sumsq: 64 fixed partials each.
// grid = (C/64, 2, 16), block = 256. Chunk = 4096 rows, 4 row-lanes/channel.
__global__ __launch_bounds__(256) void stats_partial_kernel(
    const float* __restrict__ x, float* __restrict__ ps, float* __restrict__ pss, int C)
{
  const int c  = blockIdx.x * 64 + (threadIdx.x & 63);
  const int rl = threadIdx.x >> 6;            // 0..3
  const int b  = blockIdx.y;
  const int j  = blockIdx.z * 4 + rl;         // 0..63
  const size_t rbase = (size_t)b * NPERB + (size_t)blockIdx.z * 4096;
  float s = 0.0f, ss = 0.0f;
  for (int rr = rl; rr < 4096; rr += 4) {
    float v = x[(rbase + rr) * C + c];
    s += v; ss += v * v;
  }
  const size_t o = (size_t)j * (2 * C) + (size_t)b * C + c;
  ps[o] = s; pss[o] = ss;
}

__global__ void finalize_stats_kernel(const float* __restrict__ ps, const float* __restrict__ pss,
                                      float* __restrict__ mean, float* __restrict__ rstd, int C)
{
  const int i = blockIdx.x * blockDim.x + threadIdx.x;   // over 2*C = B*C
  if (i >= 2 * C) return;
  float s = 0.0f, ss = 0.0f;
  for (int j = 0; j < 64; ++j) { s += ps[(size_t)j * 2 * C + i]; ss += pss[(size_t)j * 2 * C + i]; }
  const float inv = 1.0f / 65536.0f;
  float m = s * inv;
  float var = ss * inv - m * m;
  mean[i] = m;
  rstd[i] = rsqrtf(var + EPSV);
}

// normalize + relu -> fp16 next-layer input (writes zero pad row i==NVOX)
__global__ void norm_relu_f16_kernel(const float* __restrict__ x, const float* __restrict__ mean,
                                     const float* __restrict__ rstd, _Float16* __restrict__ oh,
                                     int logC)
{
  const size_t idx = (size_t)blockIdx.x * blockDim.x + threadIdx.x;
  const int C = 1 << logC;
  const size_t total = ((size_t)NVOX + 1) << logC;
  if (idx >= total) return;
  const size_t i = idx >> logC;
  const int c = (int)(idx & (C - 1));
  _Float16 o = (_Float16)0.0f;
  if (i < NVOX) {
    const int b = (int)(i >> 16);
    float g = (x[idx] - mean[b * C + c]) * rstd[b * C + c];
    o = (_Float16)(g > 0.0f ? g : 0.0f);
  }
  oh[idx] = o;
}

// final layer: normalize + relu in place on d_out (C = 1024)
__global__ void norm_relu_f32_kernel(float* __restrict__ x, const float* __restrict__ mean,
                                     const float* __restrict__ rstd)
{
  const size_t idx = (size_t)blockIdx.x * blockDim.x + threadIdx.x;
  if (idx >= (size_t)NVOX * 1024) return;
  const size_t i = idx >> 10;
  const int c = (int)(idx & 1023);
  const int b = (int)(i >> 16);
  float g = (x[idx] - mean[b * 1024 + c]) * rstd[b * 1024 + c];
  x[idx] = g > 0.0f ? g : 0.0f;
}

// ---------------------------------------------------------------------------
extern "C" void kernel_launch(void* const* d_in, const int* in_sizes, int n_in,
                              void* d_out, int out_size, void* d_ws, size_t ws_size,
                              hipStream_t stream) {
  (void)in_sizes; (void)n_in; (void)out_size; (void)ws_size;
  const float* feats = (const float*)d_in[0];
  const float* W1 = (const float*)d_in[1];
  const float* W2 = (const float*)d_in[2];
  const float* W3 = (const float*)d_in[3];
  const float* W4 = (const float*)d_in[4];
  const int*   neigh = (const int*)d_in[5];
  float* out = (float*)d_out;

  char* ws = (char*)d_ws;
  size_t off = 0;
  auto wsalloc = [&](size_t bytes) -> void* {
    void* p = ws + off;
    off = (off + bytes + 255) & ~(size_t)255;
    return p;
  };
  _Float16* fhA    = (_Float16*)wsalloc((size_t)(NVOX + 16) * 256 * 2);
  _Float16* fhB    = (_Float16*)wsalloc((size_t)(NVOX + 16) * 256 * 2);
  float*    conv32 = (float*)   wsalloc((size_t)NVOX * 256 * 4);
  _Float16* pw1    = (_Float16*)wsalloc((size_t)27 * 32  * 64   * 2);
  _Float16* pw2    = (_Float16*)wsalloc((size_t)27 * 64  * 128  * 2);
  _Float16* pw3    = (_Float16*)wsalloc((size_t)27 * 128 * 256  * 2);
  _Float16* pw4    = (_Float16*)wsalloc((size_t)27 * 256 * 1024 * 2);
  float*    ps     = (float*)   wsalloc((size_t)64 * 2 * 1024 * 4);
  float*    pss    = (float*)   wsalloc((size_t)64 * 2 * 1024 * 4);
  float*    meanb  = (float*)   wsalloc((size_t)2 * 1024 * 4);
  float*    rstdb  = (float*)   wsalloc((size_t)2 * 1024 * 4);

  // ---- input prep + weight repack (cheap, once per call) ----
  feats_prep_kernel<<<((NVOX + 1) * 32 + 255) / 256, 256, 0, stream>>>(feats, fhA);
  repack_w_kernel<<<dim3(27, 1, 4),  512, 0, stream>>>(W1, pw1, 3,   64);
  repack_w_kernel<<<dim3(27, 2, 8),  512, 0, stream>>>(W2, pw2, 64,  128);
  repack_w_kernel<<<dim3(27, 4, 16), 512, 0, stream>>>(W3, pw3, 128, 256);
  repack_w_kernel<<<dim3(27, 8, 64), 512, 0, stream>>>(W4, pw4, 256, 1024);

  const int convBlocksX = NVOX / (32 * 8);  // 512

  // ---- layer 1: 32(pad of 3) -> 64 ----
  conv_wmma_kernel<32, 64><<<dim3(convBlocksX, 1), 256, 0, stream>>>(fhA, pw1, neigh, conv32);
  stats_partial_kernel<<<dim3(1, 2, 16), 256, 0, stream>>>(conv32, ps, pss, 64);
  finalize_stats_kernel<<<1, 256, 0, stream>>>(ps, pss, meanb, rstdb, 64);
  norm_relu_f16_kernel<<<(int)((((size_t)NVOX + 1) * 64 + 255) / 256), 256, 0, stream>>>(
      conv32, meanb, rstdb, fhB, 6);

  // ---- layer 2: 64 -> 128 ----
  conv_wmma_kernel<64, 128><<<dim3(convBlocksX, 2), 256, 0, stream>>>(fhB, pw2, neigh, conv32);
  stats_partial_kernel<<<dim3(2, 2, 16), 256, 0, stream>>>(conv32, ps, pss, 128);
  finalize_stats_kernel<<<1, 256, 0, stream>>>(ps, pss, meanb, rstdb, 128);
  norm_relu_f16_kernel<<<(int)((((size_t)NVOX + 1) * 128 + 255) / 256), 256, 0, stream>>>(
      conv32, meanb, rstdb, fhA, 7);

  // ---- layer 3: 128 -> 256 ----
  conv_wmma_kernel<128, 256><<<dim3(convBlocksX, 4), 256, 0, stream>>>(fhA, pw3, neigh, conv32);
  stats_partial_kernel<<<dim3(4, 2, 16), 256, 0, stream>>>(conv32, ps, pss, 256);
  finalize_stats_kernel<<<2, 256, 0, stream>>>(ps, pss, meanb, rstdb, 256);
  norm_relu_f16_kernel<<<(int)((((size_t)NVOX + 1) * 256 + 255) / 256), 256, 0, stream>>>(
      conv32, meanb, rstdb, fhB, 8);

  // ---- layer 4: 256 -> 1024 (conv straight into d_out, then norm in place) ----
  conv_wmma_kernel<256, 1024><<<dim3(convBlocksX, 16), 256, 0, stream>>>(fhB, pw4, neigh, out);
  stats_partial_kernel<<<dim3(16, 2, 16), 256, 0, stream>>>(out, ps, pss, 1024);
  finalize_stats_kernel<<<8, 256, 0, stream>>>(ps, pss, meanb, rstdb, 1024);
  norm_relu_f32_kernel<<<(int)(((size_t)NVOX * 1024 + 255) / 256), 256, 0, stream>>>(
      out, meanb, rstdb);
}
